// Interactor_19078244729242
// MI455X (gfx1250) — matrix operand order repaired and checked by reading, not compiled, not run on hardware
//
#include <hip/hip_runtime.h>
#include <hip/hip_bf16.h>

// ---------------- problem constants ----------------
#define BB   64      // batch
#define TT   512     // time steps
#define NN   80      // attention slots
#define HT   512     // textual hidden
#define HV   1024    // visual hidden
#define HI   512     // ilstm hidden
#define G4   2048    // 4*HI
#define NBLK 64      // persistent-scan workgroups (one per batch)

typedef __bf16 bf16_t;
typedef __attribute__((ext_vector_type(16))) __bf16 v16bf;
typedef __attribute__((ext_vector_type(8)))  __bf16 v8bf;
typedef __attribute__((ext_vector_type(8)))  float  v8f;

// ---------------- math helpers ----------------
__device__ __forceinline__ float fast_sigmoid(float x) {
    return 1.0f / (1.0f + __expf(-x));
}
__device__ __forceinline__ float fast_tanh(float x) {
    // monotone, saturates exactly to +/-1, single v_exp_f32
    return 1.0f - 2.0f / (1.0f + __expf(2.0f * x));
}
__device__ __forceinline__ float wave_reduce_sum(float v) {
#pragma unroll
    for (int off = 16; off > 0; off >>= 1) v += __shfl_xor(v, off, 32);
    return v;
}

// ---------------- device-wide barrier (persistent kernel) ----------------
__device__ __forceinline__ void grid_barrier(unsigned* cnt, unsigned* gen) {
    __threadfence();            // make this thread's global writes visible (L2/device scope)
    __syncthreads();
    if (threadIdx.x == 0) {
        unsigned g = __hip_atomic_load(gen, __ATOMIC_RELAXED, __HIP_MEMORY_SCOPE_AGENT);
        unsigned a = __hip_atomic_fetch_add(cnt, 1u, __ATOMIC_ACQ_REL, __HIP_MEMORY_SCOPE_AGENT);
        if (a == NBLK - 1u) {
            __hip_atomic_store(cnt, 0u, __ATOMIC_RELAXED, __HIP_MEMORY_SCOPE_AGENT);
            __hip_atomic_store(gen, g + 1u, __ATOMIC_RELEASE, __HIP_MEMORY_SCOPE_AGENT);
        } else {
            while (__hip_atomic_load(gen, __ATOMIC_ACQUIRE, __HIP_MEMORY_SCOPE_AGENT) == g)
                __builtin_amdgcn_s_sleep(2);
        }
    }
    __syncthreads();
}

// ---------------- fp32 -> bf16 conversion ----------------
__global__ __launch_bounds__(256) void k_cvt_bf16(const float* __restrict__ src,
                                                  bf16_t* __restrict__ dst, size_t n) {
    size_t i = (size_t)blockIdx.x * blockDim.x + threadIdx.x;
    size_t stride = (size_t)gridDim.x * blockDim.x;
    for (; i < n; i += stride) dst[i] = (bf16_t)src[i];
}

// W_cat[j][0:512)  = W_ih[j][HV + k]  (textual columns)
// W_cat[j][512:1024) = W_hh[j][k-512]
// bsum[j] = b_ih[j] + b_hh[j]
__global__ __launch_bounds__(256) void k_build_wcat(const float* __restrict__ W_ih,
                                                    const float* __restrict__ W_hh,
                                                    const float* __restrict__ b_ih,
                                                    const float* __restrict__ b_hh,
                                                    bf16_t* __restrict__ Wcat,
                                                    float* __restrict__ bsum) {
    int j = blockIdx.x;                       // 0..2047
    for (int k = threadIdx.x; k < 1024; k += 256) {
        float v = (k < HI) ? W_ih[(size_t)j * (HT + HV) + HV + k]
                           : W_hh[(size_t)j * HI + (k - HI)];
        Wcat[(size_t)j * 1024 + k] = (bf16_t)v;
    }
    if (threadIdx.x == 0) bsum[j] = b_ih[j] + b_hh[j];
}

// ---------------- generic NT GEMM: C[M,N] = A[M,K]*W[N,K]^T + bias ----------------
// Wave computes a 16x64 tile (4 accumulators, A fragment reused 4x).
// A fragment (16x32 bf16, contiguous k order chosen here, same as B): lane l: row = l&15, half = l>>4;
//   holds K = half*16 + {0..15}  (two b128 loads)
// B fragment (32x16 bf16): lane l: col = l&15; K = (l>>4)*16 + {0..15} (one 32B load)
__global__ __launch_bounds__(256) void k_gemm_bf16_nt(
    const bf16_t* __restrict__ A, int lda,
    const bf16_t* __restrict__ W, int ldw,
    const float* __restrict__ bias,
    float* __restrict__ C, int ldc,
    int M, int N, int K) {
    int wave = (int)((blockIdx.x * blockDim.x + threadIdx.x) >> 5);
    int lane = threadIdx.x & 31;
    int ntiles = N >> 6;                          // 64-wide tiles
    int mtiles = M >> 4;
    if (wave >= mtiles * ntiles) return;
    int mt = wave / ntiles, nt = wave % ntiles;
    int r = lane & 15, half = lane >> 4;

    const bf16_t* arow = A + (size_t)(mt * 16 + r) * lda + half * 16;
    v8f acc[4];
#pragma unroll
    for (int j = 0; j < 4; ++j)
#pragma unroll
        for (int v = 0; v < 8; ++v) acc[j][v] = 0.0f;

    for (int k0 = 0; k0 < K; k0 += 32) {
        v8bf alo = *(const v8bf*)(arow + k0);
        v8bf ahi = *(const v8bf*)(arow + k0 + 8);
        v16bf af = __builtin_shufflevector(alo, ahi,
                        0, 1, 2, 3, 4, 5, 6, 7, 8, 9, 10, 11, 12, 13, 14, 15);
#pragma unroll
        for (int j = 0; j < 4; ++j) {
            const bf16_t* pb = W + (size_t)(nt * 64 + j * 16 + r) * ldw + k0 + half * 16;
            v16bf bfr = *(const v16bf*)pb;
            acc[j] = __builtin_amdgcn_wmma_f32_16x16x32_bf16(
                         false, af, false, bfr, (short)0, acc[j], false, false);
        }
    }
#pragma unroll
    for (int j = 0; j < 4; ++j) {
        int col = nt * 64 + j * 16 + r;
        float bv = bias ? bias[col] : 0.0f;
#pragma unroll
        for (int v = 0; v < 8; ++v) {
            int row = mt * 16 + v + half * 8;     // C/D layout: VGPR v -> M = v + half*8
            C[(size_t)row * ldc + col] = acc[j][v] + bv;
        }
    }
}

// ---------------- persistent sequential scan ----------------
__global__ __launch_bounds__(256) void k_scan(
    const float*  __restrict__ h_s,      // (B*N, HT) f32
    const float*  __restrict__ W_R,      // (HI, HI) f32
    const float*  __restrict__ b_R,      // (HI)
    const float*  __restrict__ W_w,      // (HI)
    const float*  __restrict__ b_w,      // (1)
    const float*  __restrict__ S_proj,   // (B*N, HI) f32  (has b_S)
    const float*  __restrict__ V_proj,   // (B*T, HI) f32  (has b_V)
    const float*  __restrict__ IHV,      // (B*T, 4HI) f32 (has b_ih+b_hh)
    const bf16_t* __restrict__ Wcat,     // (4HI, 1024) bf16
    float*  __restrict__ beta,           // (N, B) scratch
    float*  __restrict__ gates,          // (B, 4HI) scratch
    float*  __restrict__ h_state,        // (B, HI) f32
    float*  __restrict__ c_state,        // (B, HI) f32
    bf16_t* __restrict__ hblob,          // (B, 1024) bf16: [H_t_s | h_prev]
    float*  __restrict__ out,            // (B, T, HI) f32
    unsigned* __restrict__ bar) {
    __shared__ float sh_h[HI];           // 2 KB
    __shared__ float sh_rv[HI];          // 2 KB
    __shared__ float sh_alpha[NN * BB];  // 20 KB

    const int tid  = threadIdx.x;
    const int lane = tid & 31;
    const int wib  = tid >> 5;           // wave-in-block 0..7
    const int b    = blockIdx.x;         // this block owns batch b in phases AB/CD
    unsigned* cnt = bar;
    unsigned* gen = bar + 16;

    for (int t = 0; t < TT; ++t) {
        // ===== Phase AB: rv = h_prev@W_R^T + b_R + V_proj[t]; beta (no inter-block dep) =====
        for (int i = tid; i < HI; i += 256) sh_h[i] = h_state[b * HI + i];
        __syncthreads();
        const float* vrow = V_proj + ((size_t)b * TT + t) * HI;
        for (int i = tid; i < HI; i += 256) {
            const float* wr = W_R + (size_t)i * HI;
            float s = 0.0f;
#pragma unroll 8
            for (int k = 0; k < HI; ++k) s += sh_h[k] * wr[k];
            sh_rv[i] = s + b_R[i] + vrow[i];
        }
        __syncthreads();
        for (int n = wib; n < NN; n += 8) {
            const float* srow = S_proj + ((size_t)(b * NN + n)) * HI;
            float part = 0.0f;
            for (int i = lane; i < HI; i += 32)
                part += fast_tanh(srow[i] + sh_rv[i]) * W_w[i];
            part = wave_reduce_sum(part);
            if (lane == 0) beta[n * BB + b] = part + b_w[0];
        }
        grid_barrier(cnt, gen);

        // ===== Phase CD: softmax over BATCH dim (redundant per block) + H_t_s =====
        if (tid < NN) {
            int n = tid;
            float mx = -1e30f;
            for (int q = 0; q < BB; ++q) mx = fmaxf(mx, beta[n * BB + q]);
            float sum = 0.0f;
            for (int q = 0; q < BB; ++q) sum += __expf(beta[n * BB + q] - mx);
            float inv = 1.0f / sum;
            for (int q = 0; q < BB; ++q)
                sh_alpha[n * BB + q] = __expf(beta[n * BB + q] - mx) * inv;
        }
        __syncthreads();
        for (int c = tid; c < HI; c += 256) {
            float s = 0.0f;
#pragma unroll 8
            for (int n = 0; n < NN; ++n)
                s += h_s[((size_t)(b * NN + n)) * HI + c] * sh_alpha[n * BB + b];
            hblob[b * 1024 + c] = (bf16_t)s;     // first half of A row
        }
        grid_barrier(cnt, gen);

        // ===== Phase E: gates = [H_t_s|h_prev] @ Wcat^T + IHV[t]  (WMMA, 512 wave-tiles) =====
        {
            int wglob = b * 8 + wib;             // 0..511
            int mt = wglob >> 7;                 // 0..3   (M = 64)
            int nt = wglob & 127;                // 0..127 (N = 2048)
            int r = lane & 15, half = lane >> 4;
            const bf16_t* arow = hblob + (size_t)(mt * 16 + r) * 1024 + half * 16;
            const bf16_t* brow = Wcat + (size_t)(nt * 16 + r) * 1024 + half * 16;
            v8f acc;
#pragma unroll
            for (int v = 0; v < 8; ++v) acc[v] = 0.0f;
            for (int k0 = 0; k0 < 1024; k0 += 32) {
                v8bf alo = *(const v8bf*)(arow + k0);
                v8bf ahi = *(const v8bf*)(arow + k0 + 8);
                v16bf af = __builtin_shufflevector(alo, ahi,
                                0, 1, 2, 3, 4, 5, 6, 7, 8, 9, 10, 11, 12, 13, 14, 15);
                v16bf bfr = *(const v16bf*)(brow + k0);
                acc = __builtin_amdgcn_wmma_f32_16x16x32_bf16(
                          false, af, false, bfr, (short)0, acc, false, false);
            }
            int col = nt * 16 + r;
#pragma unroll
            for (int v = 0; v < 8; ++v) {
                int row = mt * 16 + v + half * 8;          // batch index
                gates[(size_t)row * G4 + col] =
                    acc[v] + IHV[((size_t)row * TT + t) * G4 + col];
            }
        }
        grid_barrier(cnt, gen);

        // ===== Phase F: LSTM pointwise =====
        for (int idx = b * 256 + tid; idx < BB * HI; idx += NBLK * 256) {
            int q = idx >> 9, i = idx & (HI - 1);
            const float* g = gates + (size_t)q * G4;
            float ig = fast_sigmoid(g[i]);
            float fg = fast_sigmoid(g[HI + i]);
            float gg = fast_tanh(g[2 * HI + i]);
            float og = fast_sigmoid(g[3 * HI + i]);
            float c = fg * c_state[idx] + ig * gg;
            float h = og * fast_tanh(c);
            c_state[idx] = c;
            h_state[idx] = h;
            hblob[q * 1024 + HI + i] = (bf16_t)h;          // second half of A row
            out[((size_t)q * TT + t) * HI + i] = h;
        }
        grid_barrier(cnt, gen);
    }
}

// ---------------- host launch ----------------
extern "C" void kernel_launch(void* const* d_in, const int* in_sizes, int n_in,
                              void* d_out, int out_size, void* d_ws, size_t ws_size,
                              hipStream_t stream) {
    (void)in_sizes; (void)n_in; (void)out_size; (void)ws_size;
    const float* h_v  = (const float*)d_in[0];
    const float* h_s  = (const float*)d_in[1];
    const float* W_S  = (const float*)d_in[2];
    const float* b_S  = (const float*)d_in[3];
    const float* W_V  = (const float*)d_in[4];
    const float* b_V  = (const float*)d_in[5];
    const float* W_R  = (const float*)d_in[6];
    const float* b_R  = (const float*)d_in[7];
    const float* W_w  = (const float*)d_in[8];
    const float* b_w  = (const float*)d_in[9];
    const float* W_ih = (const float*)d_in[10];
    const float* b_ih = (const float*)d_in[11];
    const float* W_hh = (const float*)d_in[12];
    const float* b_hh = (const float*)d_in[13];
    float* out = (float*)d_out;

    char* base = (char*)d_ws;
    size_t off = 0;
    auto alloc = [&](size_t bytes) -> void* {
        off = (off + 255) & ~(size_t)255;
        void* q = base + off;
        off += bytes;
        return q;
    };

    const size_t nHS  = (size_t)BB * NN * HT;          // 2,621,440
    const size_t nHVt = (size_t)BB * TT * HV;          // 33,554,432
    const size_t nBT  = (size_t)BB * TT;               // 32,768

    bf16_t* hs_bf  = (bf16_t*)alloc(nHS * 2);
    bf16_t* hv_bf  = (bf16_t*)alloc(nHVt * 2);
    bf16_t* WS_bf  = (bf16_t*)alloc((size_t)HI * HT * 2);
    bf16_t* WV_bf  = (bf16_t*)alloc((size_t)HI * HV * 2);
    bf16_t* Wih_bf = (bf16_t*)alloc((size_t)G4 * (HT + HV) * 2);
    bf16_t* Wcat   = (bf16_t*)alloc((size_t)G4 * 1024 * 2);
    float*  bsum   = (float*)alloc((size_t)G4 * 4);
    float*  S_proj = (float*)alloc((size_t)BB * NN * HI * 4);
    float*  V_proj = (float*)alloc(nBT * HI * 4);
    float*  IHV    = (float*)alloc(nBT * G4 * 4);
    float*  beta   = (float*)alloc((size_t)NN * BB * 4);
    float*  gates  = (float*)alloc((size_t)BB * G4 * 4);
    float*  h_st   = (float*)alloc((size_t)BB * HI * 4);
    float*  c_st   = (float*)alloc((size_t)BB * HI * 4);
    bf16_t* hblob  = (bf16_t*)alloc((size_t)BB * 1024 * 2);
    unsigned* bar  = (unsigned*)alloc(256);

    const int thr = 256;
    auto nblk = [&](size_t n) { return (unsigned)((n + thr - 1) / thr); };

    // fp32 -> bf16 packs
    k_cvt_bf16<<<nblk(nHS), thr, 0, stream>>>(h_s, hs_bf, nHS);
    k_cvt_bf16<<<nblk(nHVt), thr, 0, stream>>>(h_v, hv_bf, nHVt);
    k_cvt_bf16<<<nblk((size_t)HI * HT), thr, 0, stream>>>(W_S, WS_bf, (size_t)HI * HT);
    k_cvt_bf16<<<nblk((size_t)HI * HV), thr, 0, stream>>>(W_V, WV_bf, (size_t)HI * HV);
    k_cvt_bf16<<<nblk((size_t)G4 * (HT + HV)), thr, 0, stream>>>(W_ih, Wih_bf,
                                                                 (size_t)G4 * (HT + HV));
    k_build_wcat<<<G4, thr, 0, stream>>>(W_ih, W_hh, b_ih, b_hh, Wcat, bsum);

    // Hoisted WMMA GEMMs (16x64 wave tiles; 8 waves/block)
    // S_proj = h_s @ W_S^T + b_S      : M=5120, N=512, K=512
    k_gemm_bf16_nt<<<(5120 / 16) * (512 / 64) / 8, thr, 0, stream>>>(
        hs_bf, HT, WS_bf, HT, b_S, S_proj, HI, BB * NN, HI, HT);
    // V_proj = h_v @ W_V^T + b_V      : M=32768, N=512, K=1024
    k_gemm_bf16_nt<<<(32768 / 16) * (512 / 64) / 8, thr, 0, stream>>>(
        hv_bf, HV, WV_bf, HV, b_V, V_proj, HI, BB * TT, HI, HV);
    // IHV = h_v @ W_ih[:, :HV]^T + (b_ih + b_hh) : M=32768, N=2048, K=1024
    k_gemm_bf16_nt<<<(32768 / 16) * (2048 / 64) / 8, thr, 0, stream>>>(
        hv_bf, HV, Wih_bf, HT + HV, bsum, IHV, G4, BB * TT, G4, HV);

    // zero recurrent state + barrier
    hipMemsetAsync(h_st, 0, (size_t)BB * HI * 4, stream);
    hipMemsetAsync(c_st, 0, (size_t)BB * HI * 4, stream);
    hipMemsetAsync(hblob, 0, (size_t)BB * 1024 * 2, stream);
    hipMemsetAsync(bar, 0, 256, stream);

    k_scan<<<NBLK, thr, 0, stream>>>(h_s, W_R, b_R, W_w, b_w, S_proj, V_proj, IHV,
                                     Wcat, beta, gates, h_st, c_st, hblob, out, bar);
}
